// PMBTrueEvolving_87041807221422
// MI455X (gfx1250) — matrix operand order repaired and checked by reading, not compiled, not run on hardware
//
#include <hip/hip_runtime.h>

typedef __attribute__((ext_vector_type(16))) __bf16 v16bf;
typedef __attribute__((ext_vector_type(8)))  __bf16 v8bf;
typedef __attribute__((ext_vector_type(8)))  float  v8f;
typedef unsigned short bfraw;

#define BB   4
#define SS   2048
#define DD   1024
#define HH   16
#define HDD  64

__device__ __forceinline__ __bf16 f2bf(float f) {
  unsigned int u = __builtin_bit_cast(unsigned int, f);
  unsigned int r = (u + 0x7FFFu + ((u >> 16) & 1u)) >> 16;
  unsigned short s = (unsigned short)r;
  return __builtin_bit_cast(__bf16, s);
}

__device__ __forceinline__ v16bf mk16(v8bf lo, v8bf hi) {
  return __builtin_shufflevector(lo, hi, 0, 1, 2, 3, 4, 5, 6, 7,
                                          8, 9, 10, 11, 12, 13, 14, 15);
}

__device__ __forceinline__ v16bf ld16(const __bf16* p, int hi_off) {
  v8bf lo = *(const v8bf*)p;
  v8bf hi = *(const v8bf*)(p + hi_off);
  return mk16(lo, hi);
}

// ---- DPP16 butterfly reductions within each 16-lane half (no LDS, no waits) ----
template<int CTRL>
__device__ __forceinline__ float dpp_f(float v) {
  int x = __builtin_bit_cast(int, v);
  int y = __builtin_amdgcn_update_dpp(0, x, CTRL, 0xF, 0xF, true);
  return __builtin_bit_cast(float, y);
}
__device__ __forceinline__ float red_max16(float v) {
  v = fmaxf(v, dpp_f<0xB1>(v));    // quad_perm(1,0,3,2) == xor1
  v = fmaxf(v, dpp_f<0x4E>(v));    // quad_perm(2,3,0,1) == xor2
  v = fmaxf(v, dpp_f<0x141>(v));   // row_half_mirror   == xor4 (quads uniform)
  v = fmaxf(v, dpp_f<0x140>(v));   // row_mirror        == xor8 (8-groups uniform)
  return v;
}
__device__ __forceinline__ float red_sum16(float v) {
  v += dpp_f<0xB1>(v);
  v += dpp_f<0x4E>(v);
  v += dpp_f<0x141>(v);
  v += dpp_f<0x140>(v);
  return v;
}

// ---------------- generic C = A[M,K] * W[N,K]^T (+bias) ----------------
template<bool OUT_BF16, bool HAS_BIAS>
__global__ void __launch_bounds__(256, 1)
gemm_wmma(const bfraw* __restrict__ Araw, const bfraw* __restrict__ Braw,
          const float* __restrict__ bias, void* __restrict__ Cv,
          int M, int N, int K) {
  const __bf16* A  = (const __bf16*)Araw;
  const __bf16* Bw = (const __bf16*)Braw;
  const int lane = threadIdx.x & 31;
  const int wave = threadIdx.x >> 5;
  const int m0 = blockIdx.x << 4;
  const int n0 = (blockIdx.y << 7) + (wave << 4);
  const int r = lane & 15;
  const int koffA = (lane & 16) ? 8 : 0;    // A frag: K {0..7,16..23} / {8..15,24..31}
  const int koffB = (lane & 16) ? 8 : 0;    // B frag: K {0..7,16..23} / {8..15,24..31}
  const __bf16* ap = A  + (size_t)(m0 + r) * K + koffA;
  const __bf16* bp = Bw + (size_t)(n0 + r) * K + koffB;
  v8f acc = {0.f,0.f,0.f,0.f,0.f,0.f,0.f,0.f};
#pragma unroll 4
  for (int k = 0; k < K; k += 32) {
    v16bf a = ld16(ap + k, 16);
    v16bf b = ld16(bp + k, 16);
    acc = __builtin_amdgcn_wmma_f32_16x16x32_bf16(false, a, false, b, (short)0, acc, false, false);
  }
  const int rhi = (lane & 16) ? 8 : 0;
  const int col = n0 + r;
  float bv = 0.f;
  if (HAS_BIAS) bv = bias[col];
#pragma unroll
  for (int i = 0; i < 8; ++i) {
    float v = acc[i] + bv;
    size_t off = (size_t)(m0 + rhi + i) * N + col;
    if (OUT_BF16) ((__bf16*)Cv)[off] = f2bf(v);
    else          ((float*)Cv)[off]  = v;
  }
}

// ---------------- f32 -> bf16 convert ----------------
__global__ void cvt_bf16(const float* __restrict__ src, bfraw* __restrict__ dst, int n) {
  int i = blockIdx.x * 256 + threadIdx.x;
  if (i < n) ((__bf16*)dst)[i] = f2bf(src[i]);
}

// ---------------- top-3 of masked Gram row + softmax weights ----------------
__global__ void top3_kernel(const float* __restrict__ gram, int* __restrict__ tidx,
                            float* __restrict__ tw) {
  __shared__ float sv[256 * 3];
  __shared__ int   si[256 * 3];
  const int i = blockIdx.x;
  const int t = threadIdx.x;
  float v0 = -3.0e38f, v1 = -3.0e38f, v2 = -3.0e38f;
  int   i0 = 0, i1 = 0, i2 = 0;
  for (int j = t; j < SS; j += 256) {
    float v = (j < i) ? gram[(size_t)i * SS + j] : -1.0e9f;   // reference NEG mask
    if (v > v2) {
      if (v > v0)      { v2=v1;i2=i1; v1=v0;i1=i0; v0=v;i0=j; }
      else if (v > v1) { v2=v1;i2=i1; v1=v; i1=j; }
      else             { v2=v; i2=j; }
    }
  }
  sv[t*3]=v0; sv[t*3+1]=v1; sv[t*3+2]=v2;
  si[t*3]=i0; si[t*3+1]=i1; si[t*3+2]=i2;
  __syncthreads();
  for (int stride = 128; stride > 0; stride >>= 1) {
    if (t < stride) {
      float a0=sv[t*3],a1=sv[t*3+1],a2=sv[t*3+2];
      int   b0=si[t*3],b1=si[t*3+1],b2=si[t*3+2];
#pragma unroll
      for (int q = 0; q < 3; ++q) {
        float v = sv[(t+stride)*3+q]; int ji = si[(t+stride)*3+q];
        if (v > a2) {
          if (v > a0)      { a2=a1;b2=b1; a1=a0;b1=b0; a0=v;b0=ji; }
          else if (v > a1) { a2=a1;b2=b1; a1=v; b1=ji; }
          else             { a2=v; b2=ji; }
        }
      }
      sv[t*3]=a0;sv[t*3+1]=a1;sv[t*3+2]=a2;
      si[t*3]=b0;si[t*3+1]=b1;si[t*3+2]=b2;
    }
    __syncthreads();
  }
  if (t == 0) {
    float m = sv[0];
    float e0 = __expf(sv[0]-m), e1 = __expf(sv[1]-m), e2 = __expf(sv[2]-m);
    float inv = 1.f / (e0 + e1 + e2);
    tw[i*3]=e0*inv; tw[i*3+1]=e1*inv; tw[i*3+2]=e2*inv;
    tidx[i*3]=si[0]; tidx[i*3+1]=si[1]; tidx[i*3+2]=si[2];
  }
}

// ---------------- light sequential scan: each thread owns one column ----------------
__global__ void scan_pe(const float* __restrict__ pos_w, const float* __restrict__ pos_b,
                        const float* __restrict__ simw,
                        const int* __restrict__ tidx, const float* __restrict__ tw,
                        float* __restrict__ pe, bfraw* __restrict__ pebr) {
  __bf16* pebf = (__bf16*)pebr;
  const int d = threadIdx.x;          // 1024 threads == D
  const float pw = pos_w[d], pb = pos_b[d], sw = simw[0];
  for (int i = 0; i < SS; ++i) {
    float v = (float)i * pw + pb;
    if (i > 0) {
      int t0 = tidx[i*3+0], t1 = tidx[i*3+1], t2 = tidx[i*3+2];
      float w0 = tw[i*3+0], w1 = tw[i*3+1], w2 = tw[i*3+2];
      float acc = w0 * pe[(size_t)t0*DD + d] + w1 * pe[(size_t)t1*DD + d] + w2 * pe[(size_t)t2*DD + d];
      v += sw * acc;
    }
    pe[(size_t)i*DD + d]   = v;
    pebf[(size_t)i*DD + d] = f2bf(v);
  }
}

// ---------------- pack [hs | pos_enc] as bf16 ----------------
__global__ void pack_fused(const float* __restrict__ hs, const bfraw* __restrict__ pebr,
                           bfraw* __restrict__ far_) {
  const __bf16* pebf = (const __bf16*)pebr;
  __bf16* fa = (__bf16*)far_;
  int i = blockIdx.x * 256 + threadIdx.x;
  if (i >= BB*SS*DD) return;
  int row = i >> 10;
  int d   = i & 1023;
  int s   = row & (SS - 1);
  fa[(size_t)row * (2*DD) + d]       = f2bf(hs[i]);
  fa[(size_t)row * (2*DD) + DD + d]  = pebf[(size_t)s * DD + d];
}

// ---------------- flash attention: double-buffered async K staging ----------------
__global__ void __launch_bounds__(256, 1)
attn_flash(const bfraw* __restrict__ Qr, const bfraw* __restrict__ Kr,
           const bfraw* __restrict__ Vr, bfraw* __restrict__ Or) {
  const __bf16* Q  = (const __bf16*)Qr;
  const __bf16* Kk = (const __bf16*)Kr;
  const __bf16* V  = (const __bf16*)Vr;
  __bf16* O = (__bf16*)Or;

  __shared__ alignas(16) __bf16 sK [2][32 * 64];    // [buf][key][hd]
  __shared__ alignas(16) __bf16 sVt[2][64 * 32];    // [buf][hd][key] (transposed)
  __shared__ alignas(16) __bf16 pbuf[8 * 16 * 32];  // per-wave P tile [row][key]

  const int lane = threadIdx.x & 31;
  const int wave = threadIdx.x >> 5;
  const int b = blockIdx.x >> 4;
  const int h = blockIdx.x & 15;
  const int s0q = (blockIdx.y << 7) + (wave << 4);
  const int r = lane & 15;
  const int koffA = (lane & 16) ? 8 : 0;
  const int koffB = (lane & 16) ? 8 : 0;

  const __bf16* qp = Q + ((size_t)(b * SS + s0q + r)) * DD + h * HDD;
  v16bf qa0 = ld16(qp + koffA, 16);
  v16bf qa1 = ld16(qp + 32 + koffA, 16);

  float m[8], l[8];
  v8f zz = {0.f,0.f,0.f,0.f,0.f,0.f,0.f,0.f};
  v8f oacc[4];
#pragma unroll
  for (int i = 0; i < 8; ++i) { m[i] = -3.0e38f; l[i] = 0.f; }
#pragma unroll
  for (int c = 0; c < 4; ++c) oacc[c] = zz;

  __bf16* pmy = pbuf + wave * (16 * 32);
  const int ldrow = threadIdx.x >> 3;       // 0..31 key row
  const int seg   = (threadIdx.x & 7) << 3; // 8-elem segment

  unsigned sK_dst[2];
  sK_dst[0] = (unsigned)(unsigned long long)(&sK[0][ldrow * 64 + seg]);
  sK_dst[1] = (unsigned)(unsigned long long)(&sK[1][ldrow * 64 + seg]);
  const size_t ldbase = (size_t)(b * SS + ldrow) * DD + h * HDD + seg;
  const int NCH = SS / 32;

  // prologue: stage chunk 0 into buffer 0
  {
    const __bf16* ksrc = Kk + ldbase;
    asm volatile("global_load_async_to_lds_b128 %0, %1, off"
                 :: "v"(sK_dst[0]), "v"(ksrc) : "memory");
    v8bf vv = *(const v8bf*)(V + ldbase);
#pragma unroll
    for (int j = 0; j < 8; ++j) sVt[0][(seg + j) * 32 + ldrow] = vv[j];
    asm volatile("s_wait_asynccnt 0x0" ::: "memory");
  }

  for (int n = 0; n < NCH; ++n) {
    const int cb = n & 1;
    __syncthreads();   // buffer cb complete for all waves; buffer cb^1 free to overwrite

    if (n + 1 < NCH) { // stage chunk n+1 into buffer cb^1 while computing chunk n
      size_t nb = ldbase + (size_t)(n + 1) * 32 * DD;
      const __bf16* ksrc = Kk + nb;
      asm volatile("global_load_async_to_lds_b128 %0, %1, off"
                   :: "v"(sK_dst[cb ^ 1]), "v"(ksrc) : "memory");
      v8bf vv = *(const v8bf*)(V + nb);
#pragma unroll
      for (int j = 0; j < 8; ++j) sVt[cb ^ 1][(seg + j) * 32 + ldrow] = vv[j];
    }

#pragma unroll
    for (int kt = 0; kt < 2; ++kt) {      // two 16-key score tiles
      const __bf16* kb = &sK[cb][(kt * 16 + r) * 64 + koffB];
      v16bf b0 = ld16(kb, 16);
      v16bf b1 = ld16(kb + 32, 16);
      v8f sacc = zz;
      sacc = __builtin_amdgcn_wmma_f32_16x16x32_bf16(false, qa0, false, b0, (short)0, sacc, false, false);
      sacc = __builtin_amdgcn_wmma_f32_16x16x32_bf16(false, qa1, false, b1, (short)0, sacc, false, false);
#pragma unroll
      for (int i = 0; i < 8; ++i) {       // online softmax per row (DPP butterflies)
        float s = sacc[i] * 0.125f;       // 1/sqrt(64)
        float mnew = fmaxf(m[i], red_max16(s));
        float corr = __expf(m[i] - mnew);
        float pv   = __expf(s - mnew);
        l[i] = l[i] * corr + red_sum16(pv);
        m[i] = mnew;
        oacc[0][i] *= corr; oacc[1][i] *= corr; oacc[2][i] *= corr; oacc[3][i] *= corr;
        int rowM = ((lane & 16) ? 8 : 0) + i;
        pmy[rowM * 32 + kt * 16 + r] = f2bf(pv);
      }
    }

    { // O += P(16x32) * V(32x64); pbuf is wave-private, same-wave DS ops are in order
      v16bf pa = ld16(pmy + r * 32 + koffA, 16);
#pragma unroll
      for (int c = 0; c < 4; ++c) {
        v16bf vb = ld16(&sVt[cb][(c * 16 + r) * 32 + koffB], 16);
        oacc[c] = __builtin_amdgcn_wmma_f32_16x16x32_bf16(false, pa, false, vb, (short)0, oacc[c], false, false);
      }
    }

    // next chunk's async K copy (issued above) must land before anyone reads it
    asm volatile("s_wait_asynccnt 0x0" ::: "memory");
  }

#pragma unroll
  for (int c = 0; c < 4; ++c) {
#pragma unroll
    for (int i = 0; i < 8; ++i) {
      int rowM = ((lane & 16) ? 8 : 0) + i;
      float v = oacc[c][i] / l[i];
      O[(size_t)(b * SS + s0q + rowM) * DD + h * HDD + c * 16 + r] = f2bf(v);
    }
  }
}

__global__ void zero_f32(float* __restrict__ p, int n) {
  int i = blockIdx.x * 256 + threadIdx.x;
  if (i < n) p[i] = 0.f;
}

extern "C" void kernel_launch(void* const* d_in, const int* in_sizes, int n_in,
                              void* d_out, int out_size, void* d_ws, size_t ws_size,
                              hipStream_t stream) {
  (void)in_sizes; (void)n_in; (void)out_size; (void)ws_size;
  const float* hs        = (const float*)d_in[0];
  const float* pos_w     = (const float*)d_in[1];
  const float* pos_b     = (const float*)d_in[2];
  const float* content_w = (const float*)d_in[3];
  const float* content_b = (const float*)d_in[4];
  const float* sim_w     = (const float*)d_in[5];
  const float* fusion_w  = (const float*)d_in[6];
  const float* fusion_b  = (const float*)d_in[7];
  const float* q_w       = (const float*)d_in[8];
  const float* k_w       = (const float*)d_in[9];
  const float* v_w       = (const float*)d_in[10];
  const float* o_w       = (const float*)d_in[11];
  float* out = (float*)d_out;

  char* p = (char*)d_ws;
  auto alloc = [&](size_t bytes) { char* q = p; p += (bytes + 255) & ~(size_t)255; return q; };

  bfraw* cw_b   = (bfraw*)alloc((size_t)DD*DD*2);
  bfraw* fw_b   = (bfraw*)alloc((size_t)DD*2*DD*2);
  bfraw* qw_b   = (bfraw*)alloc((size_t)DD*DD*2);
  bfraw* kw_b   = (bfraw*)alloc((size_t)DD*DD*2);
  bfraw* vw_b   = (bfraw*)alloc((size_t)DD*DD*2);
  bfraw* ow_b   = (bfraw*)alloc((size_t)DD*DD*2);
  bfraw* hs0_b  = (bfraw*)alloc((size_t)SS*DD*2);
  bfraw* keys_b = (bfraw*)alloc((size_t)SS*DD*2);
  float* gram   = (float*)alloc((size_t)SS*SS*4);
  int*   tidx   = (int*)  alloc((size_t)SS*3*4);
  float* tw     = (float*)alloc((size_t)SS*3*4);
  float* pe     = (float*)alloc((size_t)SS*DD*4);
  bfraw* pe_b   = (bfraw*)alloc((size_t)SS*DD*2);
  bfraw* fa_b   = (bfraw*)alloc((size_t)BB*SS*2*DD*2);
  bfraw* fu_b   = (bfraw*)alloc((size_t)BB*SS*DD*2);
  bfraw* q_b    = (bfraw*)alloc((size_t)BB*SS*DD*2);
  bfraw* k_b    = (bfraw*)alloc((size_t)BB*SS*DD*2);
  bfraw* v_b    = (bfraw*)alloc((size_t)BB*SS*DD*2);
  bfraw* at_b   = (bfraw*)alloc((size_t)BB*SS*DD*2);

  cvt_bf16<<<(DD*DD+255)/256,   256, 0, stream>>>(content_w, cw_b, DD*DD);
  cvt_bf16<<<(2*DD*DD+255)/256, 256, 0, stream>>>(fusion_w,  fw_b, 2*DD*DD);
  cvt_bf16<<<(DD*DD+255)/256,   256, 0, stream>>>(q_w, qw_b, DD*DD);
  cvt_bf16<<<(DD*DD+255)/256,   256, 0, stream>>>(k_w, kw_b, DD*DD);
  cvt_bf16<<<(DD*DD+255)/256,   256, 0, stream>>>(v_w, vw_b, DD*DD);
  cvt_bf16<<<(DD*DD+255)/256,   256, 0, stream>>>(o_w, ow_b, DD*DD);
  cvt_bf16<<<(SS*DD+255)/256,   256, 0, stream>>>(hs, hs0_b, SS*DD);   // batch 0 only

  // keys = hs0 @ Wc^T + bc  (bf16 out, feeds Gram)
  gemm_wmma<true,  true ><<<dim3(SS/16, DD/128), 256, 0, stream>>>(hs0_b, cw_b, content_b, keys_b, SS, DD, DD);
  // Gram = keys @ keys^T (f32 out)
  gemm_wmma<false, false><<<dim3(SS/16, SS/128), 256, 0, stream>>>(keys_b, keys_b, nullptr, gram, SS, SS, DD);
  top3_kernel<<<SS, 256, 0, stream>>>(gram, tidx, tw);
  scan_pe<<<1, DD, 0, stream>>>(pos_w, pos_b, sim_w, tidx, tw, pe, pe_b);
  pack_fused<<<(BB*SS*DD+255)/256, 256, 0, stream>>>(hs, pe_b, fa_b);
  // fused = [hs|pe] @ Wf^T + bf
  gemm_wmma<true,  true ><<<dim3(BB*SS/16, DD/128), 256, 0, stream>>>(fa_b, fw_b, fusion_b, fu_b, BB*SS, DD, 2*DD);
  // Q,K,V projections
  gemm_wmma<true,  false><<<dim3(BB*SS/16, DD/128), 256, 0, stream>>>(fu_b, qw_b, nullptr, q_b, BB*SS, DD, DD);
  gemm_wmma<true,  false><<<dim3(BB*SS/16, DD/128), 256, 0, stream>>>(fu_b, kw_b, nullptr, k_b, BB*SS, DD, DD);
  gemm_wmma<true,  false><<<dim3(BB*SS/16, DD/128), 256, 0, stream>>>(fu_b, vw_b, nullptr, v_b, BB*SS, DD, DD);
  // attention
  attn_flash<<<dim3(BB*HH, SS/128), 256, 0, stream>>>(q_b, k_b, v_b, at_b);
  // output projection -> f32 d_out
  gemm_wmma<false, false><<<dim3(BB*SS/16, DD/128), 256, 0, stream>>>(at_b, ow_b, nullptr, out, BB*SS, DD, DD);
  // new_attention_states = zeros [B,S,H,1,1]
  zero_f32<<<(BB*SS*HH+255)/256, 256, 0, stream>>>(out + (size_t)BB*SS*DD, BB*SS*HH);
}
